// GNNPolicy_83940840833466
// MI455X (gfx1250) — matrix operand order repaired and checked by reading, not compiled, not run on hardware
//
#include <hip/hip_runtime.h>
#include <hip/hip_bf16.h>

typedef __attribute__((ext_vector_type(16))) _Float16 v16h;
typedef __attribute__((ext_vector_type(8)))  float    v8f;

#define HIDDEN 64

// ---------------------------------------------------------------- degree
__global__ void k_init_deg(float* __restrict__ deg, int n) {
  int i = blockIdx.x * blockDim.x + threadIdx.x;
  if (i < n) deg[i] = 1.0f;                       // self-loop contributes 1
}

__global__ void k_deg_scatter(const int* __restrict__ ei, float* __restrict__ deg,
                              int E) {
  int e = blockIdx.x * blockDim.x + threadIdx.x;
  if (e < E) atomicAdd(&deg[ei[E + e]], 1.0f);    // dst row of edge_index
}

__global__ void k_rsqrt_inplace(float* __restrict__ deg, int n) {
  int i = blockIdx.x * blockDim.x + threadIdx.x;
  if (i < n) {
    float d = deg[i];
    deg[i] = d > 0.0f ? rsqrtf(d) : 0.0f;
  }
}

// ---------------------------------------------------------------- x @ W1 (K=4)
__global__ void k_xw1(const float* __restrict__ x, const float* __restrict__ W1,
                      float* __restrict__ out, int n) {
  int t = blockIdx.x * blockDim.x + threadIdx.x;
  if (t >= n * HIDDEN) return;
  int node = t >> 6, c = t & 63;
  float acc = 0.0f;
#pragma unroll
  for (int k = 0; k < 4; ++k) acc += x[node * 4 + k] * W1[k * HIDDEN + c];
  out[t] = acc;
}

// ------------------------------------------------- message scatter (atomic, L2)
__global__ void k_scatter(const int* __restrict__ ei, const float* __restrict__ hpre,
                          const float* __restrict__ dis, float* __restrict__ agg,
                          int E) {
  long long t = (long long)blockIdx.x * blockDim.x + threadIdx.x;
  if (t >= (long long)E * HIDDEN) return;
  int e = (int)(t >> 6), f = (int)(t & 63);
  int s = ei[e], d = ei[E + e];
  float nrm = dis[s] * dis[d];
  atomicAdd(&agg[(long long)d * HIDDEN + f], hpre[(long long)s * HIDDEN + f] * nrm);
}

// --------------------------- finalize: agg + self-loop + bias, ReLU, pack f16
__global__ void k_finalize(const float* __restrict__ agg, const float* __restrict__ hpre,
                           const float* __restrict__ dis, const float* __restrict__ bias,
                           _Float16* __restrict__ h16, int n) {
  long long t = (long long)blockIdx.x * blockDim.x + threadIdx.x;
  if (t >= (long long)n * HIDDEN) return;
  int node = (int)(t >> 6), f = (int)(t & 63);
  float di = dis[node];
  float v = agg[t] + hpre[t] * di * di + bias[f];
  h16[t] = (_Float16)fmaxf(v, 0.0f);
}

// ---------------------------------------------------------------- h1 @ W2 WMMA
// Wave tile: 16 rows x 64 cols, K=64 (2 k-steps of 32).
// W2 pre-swizzled into LDS in per-lane fragment order (32B aligned reads feed
// the WMMA directly); nt loop rolled so at most 2 B fragments are live.
__global__ void __launch_bounds__(256, 1)
k_gemm_w2(const _Float16* __restrict__ h1, const float* __restrict__ W2,
          float* __restrict__ out, int n) {
  // sWfrag[ks][nt][lane][16 f16] : 2*4*32*16*2B = 8 KB
  __shared__ __align__(16) _Float16 sWfrag[2 * 4 * 32 * 16];
  for (int t = threadIdx.x; t < 2 * 4 * 32 * 8; t += blockDim.x) {
    int v  = t & 7;
    int ln = (t >> 3) & 31;
    int nt = (t >> 8) & 3;
    int ks = t >> 10;                 // 0..1
    int kk = ks * 32 + (v >> 2) * 16 + (ln >> 4) * 8 + (v & 3) * 2;
    int nn = nt * 16 + (ln & 15);
    sWfrag[2 * t]     = (_Float16)W2[kk * 64 + nn];
    sWfrag[2 * t + 1] = (_Float16)W2[(kk + 1) * 64 + nn];
  }
  __syncthreads();

  const int lane = threadIdx.x & 31;
  const int wave = threadIdx.x >> 5;
  const int half = lane >> 4;     // 0: K-lo / C rows 0..7, 1: K-hi / C rows 8..15
  const int l16  = lane & 15;

  int tile = blockIdx.x * (blockDim.x >> 5) + wave;
  int row  = tile * 16 + l16;
  int r    = row < n ? row : n - 1;               // clamp: keep EXEC all-ones
  const bool fullTile = (tile * 16 + 16 <= n);    // wave-uniform

  // A fragments: 16x32 f16, lane row M=l16, VGPR v -> K = (v/4)*16 + half*8 + (v%4)*2
  v16h afrag[2];
#pragma unroll
  for (int ks = 0; ks < 2; ++ks)
#pragma unroll
    for (int v = 0; v < 8; ++v) {
      int k = ks * 32 + (v >> 2) * 16 + half * 8 + (v & 3) * 2;
      afrag[ks][2 * v]     = h1[(long long)r * 64 + k];
      afrag[ks][2 * v + 1] = h1[(long long)r * 64 + k + 1];
    }

#pragma unroll 1
  for (int nt = 0; nt < 4; ++nt) {
    v8f acc = {};
#pragma unroll
    for (int ks = 0; ks < 2; ++ks) {
      const v16h* bp =
          (const v16h*)(sWfrag + (size_t)((ks * 4 + nt) * 32 + lane) * 16);
      acc = __builtin_amdgcn_wmma_f32_16x16x32_f16(false, afrag[ks], false, *bp,
                                                   (short)0, acc, false, false);
    }
    // C: VGPR v -> row v+8*half, col l16
    float* op = out + (long long)(tile * 16 + 8 * half) * 64 + nt * 16 + l16;
    if (fullTile) {                               // uniform: straight stores
#pragma unroll
      for (int v = 0; v < 8; ++v) op[v * 64] = acc[v];
    } else {
#pragma unroll
      for (int v = 0; v < 8; ++v) {
        int orow = tile * 16 + v + 8 * half;
        if (orow < n) op[v * 64] = acc[v];
      }
    }
  }
}

// ---------------------------------------------------------------- edge MLP WMMA
// edge_in = [h2[src](64) | h2[dst](64) | ea(4) | 0-pad(28)]  -> K padded to 160,
// 5 WMMA k-steps; the edge_attr tail rides k-step 4 (B = Wm1 rows 128..131 zero-
// padded, A = each lane's own 4 edge_attr values) -- no cross-lane shuffles.
__global__ void __launch_bounds__(256, 1)
k_edge_mlp(const _Float16* __restrict__ h2, const int* __restrict__ ei,
           const float* __restrict__ ea, const float* __restrict__ Wm1,
           const float* __restrict__ bm1, const float* __restrict__ Wm2,
           const float* __restrict__ bm2, float* __restrict__ out,
           int E, int tilesPerWave) {
  // sWfrag[ks][nt][lane][16 f16]  -- 5*4*32*16*2B = 20 KB
  __shared__ __align__(16) _Float16 sWfrag[5 * 4 * 32 * 16];
  __shared__ float sBm1[64], sWm2[64];

  for (int t = threadIdx.x; t < 5 * 4 * 32 * 8; t += blockDim.x) {
    int v  = t & 7;
    int ln = (t >> 3) & 31;
    int nt = (t >> 8) & 3;
    int ks = t >> 10;                 // 0..4
    int kk = ks * 32 + (v >> 2) * 16 + (ln >> 4) * 8 + (v & 3) * 2;
    int nn = nt * 16 + (ln & 15);
    sWfrag[2 * t]     = (kk < 132)     ? (_Float16)Wm1[kk * 64 + nn]       : (_Float16)0.0f;
    sWfrag[2 * t + 1] = (kk + 1 < 132) ? (_Float16)Wm1[(kk + 1) * 64 + nn] : (_Float16)0.0f;
  }
  for (int i = threadIdx.x; i < 64; i += blockDim.x) { sBm1[i] = bm1[i]; sWm2[i] = Wm2[i]; }
  __syncthreads();

  const int lane = threadIdx.x & 31;
  const int wave = threadIdx.x >> 5;
  const int half = lane >> 4;
  const int l16  = lane & 15;
  const float bm2v = bm2[0];

  const int gwave = blockIdx.x * (blockDim.x >> 5) + wave;
#pragma unroll 1
  for (int t = 0; t < tilesPerWave; ++t) {
    long long base = ((long long)gwave * tilesPerWave + t) * 16;
    if (base >= E) break;                         // uniform per wave
    const bool fullTile = (base + 16 <= E);       // uniform per wave

    long long e = base + l16;
    if (e >= E) e = E - 1;                        // clamp: EXEC stays all-ones
    int s = ei[e], d = ei[E + e];

    // A fragments: k-steps 0,1 from h2[src], 2,3 from h2[dst]
    v16h afrag[4];
#pragma unroll
    for (int ks = 0; ks < 4; ++ks) {
      const _Float16* srcp = (ks < 2) ? (h2 + (long long)s * 64) : (h2 + (long long)d * 64);
      int ko = (ks & 1) * 32;
#pragma unroll
      for (int v = 0; v < 8; ++v) {
        int k = ko + (v >> 2) * 16 + half * 8 + (v & 3) * 2;
        afrag[ks][2 * v]     = srcp[k];
        afrag[ks][2 * v + 1] = srcp[k + 1];
      }
    }
    // k-step 4: K=128..131 hold this lane's edge_attr (layout: half-0 lanes own
    // K 128..135 -> elements 0..3 = ea0..ea3; half-1 lanes start at K=136 -> 0).
    v16h a4 = {};
    {
      float ea0 = ea[e * 4 + 0], ea1 = ea[e * 4 + 1];
      float ea2 = ea[e * 4 + 2], ea3 = ea[e * 4 + 3];
      a4[0] = (half == 0) ? (_Float16)ea0 : (_Float16)0.0f;
      a4[1] = (half == 0) ? (_Float16)ea1 : (_Float16)0.0f;
      a4[2] = (half == 0) ? (_Float16)ea2 : (_Float16)0.0f;
      a4[3] = (half == 0) ? (_Float16)ea3 : (_Float16)0.0f;
    }

    float rsum[8];
#pragma unroll
    for (int v = 0; v < 8; ++v) rsum[v] = 0.0f;

#pragma unroll 1
    for (int nt = 0; nt < 4; ++nt) {
      v8f acc = {};
#pragma unroll
      for (int ks = 0; ks < 4; ++ks) {
        const v16h* bp =
            (const v16h*)(sWfrag + (size_t)((ks * 4 + nt) * 32 + lane) * 16);
        acc = __builtin_amdgcn_wmma_f32_16x16x32_f16(false, afrag[ks], false, *bp,
                                                     (short)0, acc, false, false);
      }
      {
        const v16h* bp =
            (const v16h*)(sWfrag + (size_t)((4 * 4 + nt) * 32 + lane) * 16);
        acc = __builtin_amdgcn_wmma_f32_16x16x32_f16(false, a4, false, *bp,
                                                     (short)0, acc, false, false);
      }
      int nn = nt * 16 + l16;
      float b1v = sBm1[nn], w2v = sWm2[nn];
#pragma unroll
      for (int v = 0; v < 8; ++v) {
        float z = fmaxf(acc[v] + b1v, 0.0f);      // relu(edge_in@Wm1 + bm1)
        rsum[v] += z * w2v;                       // fold in z @ Wm2
      }
    }
    // reduce over the 16 columns held within each wave-half
#pragma unroll
    for (int m = 1; m < 16; m <<= 1)
#pragma unroll
      for (int v = 0; v < 8; ++v) rsum[v] += __shfl_xor(rsum[v], m, 32);

    if (l16 == 0) {
      float* op = out + base + 8 * half;
      if (fullTile) {
#pragma unroll
        for (int v = 0; v < 8; ++v) op[v] = rsum[v] + bm2v;
      } else {
#pragma unroll
        for (int v = 0; v < 8; ++v) {
          long long orow = base + v + 8 * half;
          if (orow < E) op[v] = rsum[v] + bm2v;
        }
      }
    }
  }
}

// ================================================================ host launch
extern "C" void kernel_launch(void* const* d_in, const int* in_sizes, int n_in,
                              void* d_out, int out_size, void* d_ws, size_t ws_size,
                              hipStream_t stream) {
  const float* x    = (const float*)d_in[0];
  const int*   ei   = (const int*)d_in[1];
  const float* ea   = (const float*)d_in[2];
  const float* W1   = (const float*)d_in[3];
  const float* b1   = (const float*)d_in[4];
  const float* W2   = (const float*)d_in[5];
  const float* b2   = (const float*)d_in[6];
  const float* Wm1  = (const float*)d_in[7];
  const float* bm1  = (const float*)d_in[8];
  const float* Wm2  = (const float*)d_in[9];
  const float* bm2  = (const float*)d_in[10];
  float* out = (float*)d_out;

  const int N = in_sizes[0] / 4;       // x: [N,4]
  const int E = in_sizes[2] / 4;       // edge_attr: [E,4]

  // workspace carve-out
  char* ws = (char*)d_ws;
  size_t o = 0;
  auto take = [&](size_t bytes) { char* p = ws + o; o = (o + bytes + 255) & ~(size_t)255; return p; };
  float*     dis  = (float*)take((size_t)N * 4);
  float*     bufA = (float*)take((size_t)N * HIDDEN * 4);   // h_pre (f32)
  float*     bufB = (float*)take((size_t)N * HIDDEN * 4);   // aggregation
  _Float16*  h1h  = (_Float16*)take((size_t)N * HIDDEN * 2);
  _Float16*  h2h  = (_Float16*)take((size_t)N * HIDDEN * 2);
  (void)ws_size; (void)n_in; (void)out_size;

  const int B = 256;
  const int gN   = (N + B - 1) / B;
  const int gE   = (E + B - 1) / B;
  const long long nh = (long long)N * HIDDEN;
  const long long eh = (long long)E * HIDDEN;
  const int gNH = (int)((nh + B - 1) / B);
  const int gEH = (int)((eh + B - 1) / B);

  // degrees & symmetric norm
  k_init_deg<<<gN, B, 0, stream>>>(dis, N);
  k_deg_scatter<<<gE, B, 0, stream>>>(ei, dis, E);
  k_rsqrt_inplace<<<gN, B, 0, stream>>>(dis, N);

  // layer 1: x@W1 -> scatter -> finalize(relu) -> h1 (f16)
  k_xw1<<<gNH, B, 0, stream>>>(x, W1, bufA, N);
  hipMemsetAsync(bufB, 0, (size_t)N * HIDDEN * 4, stream);
  k_scatter<<<gEH, B, 0, stream>>>(ei, bufA, dis, bufB, E);
  k_finalize<<<gNH, B, 0, stream>>>(bufB, bufA, dis, b1, h1h, N);

  // layer 2: h1@W2 (WMMA) -> scatter -> finalize(relu) -> h2 (f16)
  {
    int tiles  = (N + 15) / 16;
    int blocks = (tiles + 7) / 8;      // 8 waves (wave32) per 256-thread block
    k_gemm_w2<<<blocks, B, 0, stream>>>(h1h, W2, bufA, N);
  }
  hipMemsetAsync(bufB, 0, (size_t)N * HIDDEN * 4, stream);
  k_scatter<<<gEH, B, 0, stream>>>(ei, bufA, dis, bufB, E);
  k_finalize<<<gNH, B, 0, stream>>>(bufB, bufA, dis, b2, h2h, N);

  // edge MLP (WMMA): 16 edges/tile, 8 tiles per wave
  {
    const int TPW = 8;
    long long tiles  = ((long long)E + 15) / 16;
    long long waves  = (tiles + TPW - 1) / TPW;
    int blocks = (int)((waves + 7) / 8);
    k_edge_mlp<<<blocks, B, 0, stream>>>(h2h, ei, ea, Wm1, bm1, Wm2, bm2, out, E, TPW);
  }
}
